// CIN_81037442941666
// MI455X (gfx1250) — matrix operand order repaired and checked by reading, not compiled, not run on hardware
//
#include <hip/hip_runtime.h>

// CIN (xDeepFM) on MI455X / gfx1250, wave32 + WMMA f16->f32.
// B=1024, F=39, D=16, HIDDEN=[128,128,128].
// Each layer: cur[b,l,d] = relu( sum_{f,g} x[b,f,d]*h[b,g,d]*W[f,g,l] + bias[l] )
// => GEMM with M=B*D=16384 (one 16-row tile per batch elem), N=128, K=F0*Fk.
// Layer 0 is zero-padded from Fk=39 to Fk=64 so all layers share one code path
// (K = 39*64 = 2496 = 78 chunks of 32).
//
// FINAL (v4): h kept in LDS as f16 in [d][g] layout, so per-layer hq2 values
// are direct half2 ds_loads (no f32 copies live across the unrolled loop, no
// per-chunk cvt remat). A-frag build is exactly 8 v_pk_mul_f16 per chunk
// feeding two back-to-back WMMAs (one A reused by two N-tiles per wave).
// B-frags are pre-swizzled to the exact WMMA register layout by a prep kernel
// so each frag is one contiguous 32B load per lane.
// Note: __launch_bounds__(128, N>1) was tried and REJECTED — it forces
// scratch spills in the hot loop (worse than the VGPR-MSB churn it removes).

typedef __attribute__((ext_vector_type(16))) _Float16 v16h;
typedef __attribute__((ext_vector_type(2)))  _Float16 v2h;
typedef __attribute__((ext_vector_type(8)))  float    v8f;

#define NB    1024
#define F0    39
#define DEMB  16
#define LOUT  128
#define FK    64
#define NCH   78                       // 39*64/32 K-chunks
#define HS16  66                       // LDS halfs per d-row (64 + pad, even)
#define LAYER_HALFS (NCH * 8 * 32 * 16)  // prepped f16 weights per layer = 319488

// ---------------------------------------------------------------------------
// Prep: convert W (f32, row-major [K,128]) into f16 in WMMA B-frag layout:
//   frag element e at lane L of tile (chunk c, n-tile t) holds
//   B[K = c*32 + (e&7) + 16*(e>>3) + 8*(L/16)][N = t*16 + L%16]
// stored contiguously so each lane loads its 16 halfs with one 32B load.
// For layer 0, k = f*64 + g maps to W0[f*39 + g] (zero for g >= 39).
// ---------------------------------------------------------------------------
__global__ __launch_bounds__(256)
void cin_prep_w(const float* __restrict__ W, int isLayer0,
                _Float16* __restrict__ dst) {
  int idx  = blockIdx.x * 256 + threadIdx.x;     // exact grid, no OOB
  int e    = idx & 15;
  int lane = (idx >> 4) & 31;
  int t    = (idx >> 9) & 7;
  int c    = idx >> 12;                          // 0..77
  int k    = c * 32 + (e & 7) + ((e >> 3) << 4) + ((lane >> 4) << 3);    // 0..2495
  int n    = t * 16 + (lane & 15);
  float v;
  if (isLayer0) {
    int f = k >> 6, g = k & 63;
    v = (g < F0) ? W[(f * F0 + g) * LOUT + n] : 0.0f;
  } else {
    v = W[k * LOUT + n];
  }
  dst[idx] = (_Float16)v;
}

// ---------------------------------------------------------------------------
// Main fused kernel: one workgroup (4 waves, 128 threads) per batch element.
// Wave w owns tiles t0 = w (columns 0..63: h side) and t1 = w+4 (columns
// 64..127: direct side). Lane L: d = L%16, hf = L/16.
// A-frag element e (16x32 f16 layout): K = (e&7) + 16*(e>>3) + 8*hf, M = d.
//   => frag half2 slot p equals {x[f],x[f]} * {h[g0], h[g0+1]} with
//      f = c>>1, pair q = (c&1)*8 + p  (all compile-time after unroll).
// C-frag f32: element r -> M = r + 8*hf, N = L%16.
// ---------------------------------------------------------------------------
__global__ __launch_bounds__(128)
void cin_kernel(const float* __restrict__ x,
                const float* __restrict__ bias0,
                const float* __restrict__ bias1,
                const float* __restrict__ bias2,
                const _Float16* __restrict__ w0,
                const _Float16* __restrict__ w1,
                const _Float16* __restrict__ w2,
                float* __restrict__ out) {
  // h matrix as f16, [d][g] layout: row stride 66 halfs (132B, conflict-free)
  __shared__ _Float16 lds_h16[DEMB * HS16];

  const int b    = blockIdx.x;
  const int tid  = threadIdx.x;
  const int lane = tid & 31;
  const int wave = tid >> 5;                     // 0..3
  const int d    = lane & 15;
  const int hf   = lane >> 4;
  const int t0   = wave;                         // h tile      (l = 0..63)
  const int t1   = wave + 4;                     // direct tile (l = 64..127)

  // lds_h16 <- f16(x), zero-padded to 64 fields (padding MUST be 0: WMMA
  // propagates NaN even against zero weights).
  for (int i = tid; i < FK * DEMB; i += 128) {
    int g = i >> 4, dd = i & 15;                 // dd fastest -> coalesced x
    float v = (g < F0) ? x[(b * F0 + g) * DEMB + dd] : 0.0f;
    lds_h16[dd * HS16 + g] = (_Float16)v;
  }

  // per-lane x row, pre-converted to broadcast half2 {x,x} (A-side operand)
  v2h xp[F0];
#pragma unroll
  for (int f = 0; f < F0; ++f) {
    _Float16 t = (_Float16)x[(b * F0 + f) * DEMB + d];
    v2h tt = {t, t};
    xp[f] = tt;
  }

  __syncthreads();

  union AFrag { v16h v; v2h p[8]; };

#pragma unroll 1
  for (int li = 0; li < 3; ++li) {
    const _Float16* wb = (li == 0) ? w0 : (li == 1) ? w1 : w2;
    const float*    bp = (li == 0) ? bias0 : (li == 1) ? bias1 : bias2;

    // h values this lane-half uses, as half2 pairs straight from LDS:
    //   hq2[q] = { h[g0], h[g0+1] },  j0 = 2q,
    //   g0 = (j0>>4)*32 + ((j0>>3)&1)*16 + (j0&7) + 8*hf   (g0 even)
    v2h hq2[16];
#pragma unroll
    for (int q = 0; q < 16; ++q) {
      int j0 = 2 * q;
      int g0 = ((j0 >> 4) << 5) + (((j0 >> 3) & 1) << 4) + (j0 & 7) + (hf << 3);
      hq2[q] = *(const v2h*)&lds_h16[d * HS16 + g0];
    }
    __syncthreads();   // all reads of lds_h16 done before anyone rewrites it

    // A-frag build: 8 packed f16 muls, nothing else
    auto buildA = [&](int c) {
      AFrag af;
      const int base = (c & 1) << 3;
      const v2h xv = xp[c >> 1];
#pragma unroll
      for (int p = 0; p < 8; ++p) af.p[p] = xv * hq2[base + p];
      return af.v;
    };
    auto loadB = [&](int c, int t) {
      return *(const v16h*)(wb + ((((c * 8 + t) * 32) + lane) << 4));
    };

    v8f acc0 = {};
    v8f acc1 = {};

    // depth-1 rotation: A(c+1)/B(c+1) issued before WMMA(c)
    v16h afc = buildA(0);
    v16h bf0 = loadB(0, t0);
    v16h bf1 = loadB(0, t1);
#pragma unroll
    for (int c = 0; c < NCH; ++c) {
      v16h afn = (c + 1 < NCH) ? buildA(c + 1)    : afc;
      v16h bn0 = (c + 1 < NCH) ? loadB(c + 1, t0) : bf0;
      v16h bn1 = (c + 1 < NCH) ? loadB(c + 1, t1) : bf1;
      acc0 = __builtin_amdgcn_wmma_f32_16x16x32_f16(
          false, afc, false, bf0, (short)0, acc0, false, false);
      acc1 = __builtin_amdgcn_wmma_f32_16x16x32_f16(
          false, afc, false, bf1, (short)0, acc1, false, false);
      afc = afn; bf0 = bn0; bf1 = bn1;
    }

    const int l0 = t0 * 16 + (lane & 15);        // 0..63
    const int l1 = t1 * 16 + (lane & 15);        // 64..127
    float bl0 = bp[l0];
    float bl1 = bp[l1];

    float cres[8];
    float psum0 = 0.0f, psum1 = 0.0f;
#pragma unroll
    for (int r = 0; r < 8; ++r) {
      float v0 = acc0[r] + bl0;
      v0 = v0 > 0.0f ? v0 : 0.0f;                // relu
      cres[r] = v0;
      psum0 += v0;
      float v1 = acc1[r] + bl1;
      v1 = v1 > 0.0f ? v1 : 0.0f;
      psum1 += v1;
    }

    if (li < 2) {
      // tile0 -> h_next = cur[:, :64] -> lds_h16[d][l0] (f16), d = 8*hf + r
#pragma unroll
      for (int r = 0; r < 8; ++r)
        lds_h16[((hf << 3) + r) * HS16 + l0] = (_Float16)cres[r];
      // tile1 -> direct = cur[:, 64:] summed over d -> out cols li*64 + (l1-64)
      float tot1 = psum1 + __shfl_xor(psum1, 16, 32);
      if (lane < 16) out[b * 256 + li * 64 + (l1 - 64)] = tot1;
      __syncthreads();                           // h writes visible next layer
    } else {
      // layer 2: all 128 columns are direct -> out cols 128..255
      float tot0 = psum0 + __shfl_xor(psum0, 16, 32);
      float tot1 = psum1 + __shfl_xor(psum1, 16, 32);
      if (lane < 16) {
        out[b * 256 + 128 + l0] = tot0;
        out[b * 256 + 128 + l1] = tot1;
      }
    }
  }
}

// ---------------------------------------------------------------------------
// Launch. Inputs: x, W0, W1, W2, b0, b1, b2 (all f32). Out: (1024,256) f32.
// Workspace: 3 * 319488 f16 = ~1.83 MB of prepped weights (re-built each call:
// deterministic, no caching).
// ---------------------------------------------------------------------------
extern "C" void kernel_launch(void* const* d_in, const int* in_sizes, int n_in,
                              void* d_out, int out_size, void* d_ws, size_t ws_size,
                              hipStream_t stream) {
  const float* x  = (const float*)d_in[0];
  const float* W0 = (const float*)d_in[1];
  const float* W1 = (const float*)d_in[2];
  const float* W2 = (const float*)d_in[3];
  const float* b0 = (const float*)d_in[4];
  const float* b1 = (const float*)d_in[5];
  const float* b2 = (const float*)d_in[6];
  float* out = (float*)d_out;

  _Float16* wp0 = (_Float16*)d_ws;
  _Float16* wp1 = wp0 + LAYER_HALFS;
  _Float16* wp2 = wp1 + LAYER_HALFS;

  const int prepBlocks = LAYER_HALFS / 256;      // 1248
  cin_prep_w<<<prepBlocks, 256, 0, stream>>>(W0, 1, wp0);
  cin_prep_w<<<prepBlocks, 256, 0, stream>>>(W1, 0, wp1);
  cin_prep_w<<<prepBlocks, 256, 0, stream>>>(W2, 0, wp2);

  cin_kernel<<<NB, 128, 0, stream>>>(x, b0, b1, b2, wp0, wp1, wp2, out);
}
